// PointSIS_Feature_Extractor_3874060501269
// MI455X (gfx1250) — compile-verified, matrix operand order checked by reading, not run on hardware
//
#include <hip/hip_runtime.h>
#include <hip/hip_bf16.h>

// ---------------------------------------------------------------------------
// PointSIS feature extractor forward for MI455X (gfx1250, wave32, WMMA).
// Big GEMMs: v_wmma_f32_16x16x32_f16, 8 waves/block, 64x64 block tile,
// K-step 64 -> 4 WMMAs per barrier pair per wave. Register-buffered global
// staging (one load clause per tile), transposed-in-LDS B so all fragments
// are contiguous ds_load_b128s. Requires M%64==0, N%64==0, K%64==0
// (shapes padded where needed).
// ---------------------------------------------------------------------------

#define B_   2
#define G_   1024
#define O_   2
#define DM   256          // d_model
#define N_   (B_ * G_)    // 2048 points
#define L_   (O_ * G_)    // 2048 sequence length
#define DI   512          // d_inner
#define DS   16           // d_state
#define DC   4            // d_conv
#define DTR  16           // dt_rank
#define DBCW 48           // dt_rank + 2*d_state (logical)
#define DBCP 64           // padded row stride for dbc / x_proj out

typedef __attribute__((ext_vector_type(16))) _Float16 v16h;
typedef __attribute__((ext_vector_type(8)))  _Float16 v8h;
typedef __attribute__((ext_vector_type(8)))  float    v8f;

// -------------------------------- helpers ----------------------------------

__device__ __forceinline__ float gelu_f(float x) {
    return 0.5f * x * (1.0f + erff(x * 0.70710678118654752f));
}
__device__ __forceinline__ float silu_f(float x) {
    return x / (1.0f + expf(-x));
}
__device__ __forceinline__ float softplus_f(float x) {
    return (x > 20.0f) ? x : log1pf(expf(x));
}

// ACT codes: 0=none 1=gelu 3=softplus
template <int ACT>
__device__ __forceinline__ float apply_act(float v) {
    if (ACT == 1) return gelu_f(v);
    if (ACT == 3) return softplus_f(v);
    return v;
}

// ------------------------- WMMA tiled GEMM (f32 io) ------------------------
// C[M,N] = act(A[M,K] @ W[K,N] + bias).  REQUIRES M%64==0, N%64==0, K%64==0.
// Block: 256 threads = 8 waves. Block tile 64x64, K step 64.
// Wave w: rows (w&1)*32..+31, cols (w>>1)*16..+15 -> two 16x16 WMMA tiles
// sharing each B fragment; two K-halves per staged tile -> 4 WMMAs/iter.

template <int ACT>
__global__ __launch_bounds__(256)
void gemm_wmma_kernel(const float* __restrict__ A, int lda,
                      const float* __restrict__ W, int ldw,
                      const float* __restrict__ bias,
                      float* __restrict__ C, int ldc,
                      int M, int N, int K)
{
    __shared__ __align__(16) _Float16 As[64][72];  // [m][k], 144B rows
    __shared__ __align__(16) _Float16 Wt[64][72];  // [n][k] transposed

    const int tid  = threadIdx.x;
    const int wave = tid >> 5;
    const int lane = tid & 31;
    const int m0 = blockIdx.y * 64;
    const int n0 = blockIdx.x * 64;
    const int wr = (wave & 1) * 32;   // wave row offset (2 subtiles)
    const int wc = (wave >> 1) * 16;  // wave col offset
    const int hi = lane >> 4;         // lane half
    const int lm = lane & 15;
    const int kb = hi * 8;            // k-chunk base within fragment

    // loop-invariant staging coordinates
    const int asr = tid >> 5;         // A tile row base (step 8)
    const int asc = (tid & 31) << 1;  // A tile k (even)
    const int wsc = tid & 63;         // W tile n
    const int wsr = (tid >> 6) << 1;  // W tile k base (step 8)

    const float* aPtr = A + (size_t)(m0 + asr) * lda + asc;
    const float* wPtr = W + (size_t)wsr * ldw + n0 + wsc;
    const size_t aRow8 = (size_t)8 * lda;
    const size_t wRow8 = (size_t)8 * ldw;

    v8f acc0 = {};
    v8f acc1 = {};

    for (int kk = 0; kk < K; kk += 64) {
        // ---- gather whole tile into registers first (clause-able loads)
        float2 av[8];
        float  w0[8], w1[8];
        #pragma unroll
        for (int i = 0; i < 8; ++i)
            av[i] = *(const float2*)(aPtr + (size_t)i * aRow8);
        #pragma unroll
        for (int i = 0; i < 8; ++i) {
            w0[i] = wPtr[(size_t)i * wRow8];
            w1[i] = wPtr[(size_t)i * wRow8 + ldw];
        }
        // ---- convert + packed b32 stores to LDS
        #pragma unroll
        for (int i = 0; i < 8; ++i) {
            union { _Float16 h[2]; unsigned u; } pa, pw;
            pa.h[0] = (_Float16)av[i].x;
            pa.h[1] = (_Float16)av[i].y;
            *(unsigned*)&As[asr + 8 * i][asc] = pa.u;
            pw.h[0] = (_Float16)w0[i];
            pw.h[1] = (_Float16)w1[i];
            *(unsigned*)&Wt[wsc][wsr + 8 * i] = pw.u;
        }
        aPtr += 64;
        wPtr += (size_t)64 * ldw;
        __syncthreads();

        // ---- two K-halves, fragments as contiguous 16B chunks
        #pragma unroll
        for (int q = 0; q < 2; ++q) {
            const int ko = 32 * q + kb;
            const v8h a00 = *(const v8h*)&As[wr + lm][ko];
            const v8h a01 = *(const v8h*)&As[wr + lm][16 + ko];
            const v8h a10 = *(const v8h*)&As[wr + 16 + lm][ko];
            const v8h a11 = *(const v8h*)&As[wr + 16 + lm][16 + ko];
            const v8h b0  = *(const v8h*)&Wt[wc + lm][ko];
            const v8h b1  = *(const v8h*)&Wt[wc + lm][16 + ko];
            v16h af0 = __builtin_shufflevector(a00, a01, 0,1,2,3,4,5,6,7,8,9,10,11,12,13,14,15);
            v16h af1 = __builtin_shufflevector(a10, a11, 0,1,2,3,4,5,6,7,8,9,10,11,12,13,14,15);
            v16h bf  = __builtin_shufflevector(b0,  b1,  0,1,2,3,4,5,6,7,8,9,10,11,12,13,14,15);
            acc0 = __builtin_amdgcn_wmma_f32_16x16x32_f16(
                       false, af0, false, bf, (short)0, acc0, false, false);
            acc1 = __builtin_amdgcn_wmma_f32_16x16x32_f16(
                       false, af1, false, bf, (short)0, acc1, false, false);
        }
        __syncthreads();
    }

    // ---- store: C/D layout -> VGPR r holds row (r + 8*hi), col = lane%16
    const int gn = n0 + wc + lm;
    const float bv = bias ? bias[gn] : 0.0f;
    #pragma unroll
    for (int r = 0; r < 8; ++r) {
        const int gm = m0 + wr + r + 8 * hi;
        C[(size_t)gm * ldc + gn]        = apply_act<ACT>(acc0[r] + bv);
        C[(size_t)(gm + 16) * ldc + gn] = apply_act<ACT>(acc1[r] + bv);
    }
}

// ----------------------- tiny-K naive GEMM (K=3/4/16) ----------------------

template <int ACT>
__global__ __launch_bounds__(256)
void gemm_small_kernel(const float* __restrict__ A, int lda,
                       const float* __restrict__ W, int ldw,
                       const float* __restrict__ bias,
                       float* __restrict__ C, int ldc,
                       int M, int N, int K)
{
    const int n = blockIdx.x * 16 + (threadIdx.x & 15);
    const int m = blockIdx.y * 16 + (threadIdx.x >> 4);
    if (m >= M || n >= N) return;
    float acc = bias ? bias[n] : 0.0f;
    for (int k = 0; k < K; ++k)
        acc = fmaf(A[(size_t)m * lda + k], W[(size_t)k * ldw + n], acc);
    C[(size_t)m * ldc + n] = apply_act<ACT>(acc);
}

// ----------------------------- LayerNorm (+GELU) ---------------------------

template <int DO_GELU>
__global__ __launch_bounds__(256)
void ln_kernel(const float* __restrict__ x,
               const float* __restrict__ g, const float* __restrict__ b,
               float* __restrict__ y, int width)
{
    __shared__ float s1[256];
    __shared__ float s2[256];
    const int row = blockIdx.x;
    const int t = threadIdx.x;
    float v = (t < width) ? x[(size_t)row * width + t] : 0.0f;
    s1[t] = v;
    s2[t] = v * v;
    __syncthreads();
    #pragma unroll
    for (int off = 128; off > 0; off >>= 1) {
        if (t < off) { s1[t] += s1[t + off]; s2[t] += s2[t + off]; }
        __syncthreads();
    }
    const float mean = s1[0] / (float)width;
    const float var  = s2[0] / (float)width - mean * mean;
    const float inv  = rsqrtf(var + 1e-5f);
    if (t < width) {
        float o = (v - mean) * inv * g[t] + b[t];
        if (DO_GELU) o = gelu_f(o);
        y[(size_t)row * width + t] = o;
    }
}

// ------------------------------ misc elementwise ---------------------------

__global__ void cat_fill_kernel(const float* __restrict__ soi,
                                const float* __restrict__ w,
                                const float* __restrict__ b,
                                float* __restrict__ conc)
{
    int idx = blockIdx.x * blockDim.x + threadIdx.x;  // N_ * 128
    if (idx >= N_ * 128) return;
    int n = idx >> 7, j = idx & 127;
    conc[(size_t)n * 384 + 256 + j] = soi[n / G_] * w[j] + b[j];
}

__global__ void add_kernel(const float* __restrict__ a,
                           const float* __restrict__ b,
                           float* __restrict__ o, int n)
{
    int i = blockIdx.x * blockDim.x + threadIdx.x;
    if (i < n) o[i] = a[i] + b[i];
}

// zero-pad x_proj weight [DI,48] -> [DI,64]
__global__ void pad_xproj_kernel(const float* __restrict__ w,
                                 float* __restrict__ wp)
{
    int idx = blockIdx.x * blockDim.x + threadIdx.x;  // DI * 64
    if (idx >= DI * DBCP) return;
    int k = idx >> 6, n = idx & 63;
    wp[idx] = (n < DBCW) ? w[k * DBCW + n] : 0.0f;
}

// h[N,DM] -> scatter by order, reshaped to x_seq[B, L, DM]
__global__ void perm_fwd_kernel(const float* __restrict__ h,
                                const int* __restrict__ order,
                                float* __restrict__ xseq)
{
    int idx = blockIdx.x * blockDim.x + threadIdx.x;  // O_*N_*DM
    if (idx >= O_ * N_ * DM) return;
    int d = idx & (DM - 1);
    int r = idx / DM;
    int n = r % N_;
    int o = r / N_;
    int p = order[o * N_ + n];            // destination slot (permutation)
    int bb = p / G_, g = p % G_;
    xseq[((size_t)(bb * L_ + o * G_ + g)) * DM + d] = h[(size_t)n * DM + d];
}

// seq[B,L,DM] -> inverse scatter -> cat_o[N, O_*DM]
__global__ void perm_inv_kernel(const float* __restrict__ seq,
                                const int* __restrict__ inverse,
                                float* __restrict__ cato)
{
    int idx = blockIdx.x * blockDim.x + threadIdx.x;  // O_*N_*DM
    if (idx >= O_ * N_ * DM) return;
    int d = idx & (DM - 1);
    int r = idx / DM;
    int n = r % N_;
    int o = r / N_;
    int bb = n / G_, g = n % G_;
    float v = seq[((size_t)(bb * L_ + o * G_ + g)) * DM + d];
    int m = inverse[o * N_ + n];
    cato[(size_t)m * (O_ * DM) + o * DM + d] = v;
}

// causal depthwise conv1d (kernel 4) over xi = xz[:, :DI], fused SiLU
__global__ void conv_silu_kernel(const float* __restrict__ xz,
                                 const float* __restrict__ cw,
                                 const float* __restrict__ cb,
                                 float* __restrict__ xc)
{
    int idx = blockIdx.x * blockDim.x + threadIdx.x;  // B_*L_*DI
    if (idx >= B_ * L_ * DI) return;
    int d = idx & (DI - 1);
    int r = idx / DI;
    int l = r % L_;
    int bb = r / L_;
    float acc = cb[d];
    #pragma unroll
    for (int k = 0; k < DC; ++k) {
        int ls = l - (DC - 1) + k;
        if (ls >= 0)
            acc = fmaf(xz[((size_t)(bb * L_ + ls)) * (2 * DI) + d],
                       cw[d * DC + k], acc);
    }
    xc[idx] = silu_f(acc);
}

// ------------------------- Mamba selective scan ----------------------------
// grid (DI/16, B_), 256 threads: thread = (channel 0..15, state 0..15).
// Register-resident recurrence per (b,d,s) chain; Sum_s via wave32 xor
// shuffles inside 16-lane groups; fused skip (xc*D) and SiLU(z) gating.

__global__ __launch_bounds__(256)
void scan_kernel(const float* __restrict__ dt,    // [B*L, DI] softplus'ed
                 const float* __restrict__ xc,    // [B*L, DI]
                 const float* __restrict__ dbc,   // [B*L, DBCP] (B at +16, C at +32)
                 const float* __restrict__ xz,    // [B*L, 2*DI] (z at +DI)
                 const float* __restrict__ A_log, // [DI, DS]
                 const float* __restrict__ Dp,    // [DI]
                 float* __restrict__ y)           // [B*L, DI]
{
    const int s  = threadIdx.x & 15;
    const int d  = blockIdx.x * 16 + (threadIdx.x >> 4);
    const int bb = blockIdx.y;

    const float Aneg = -expf(A_log[d * DS + s]);
    const float Dv = Dp[d];
    float h = 0.0f;

    for (int l = 0; l < L_; ++l) {
        const size_t row = (size_t)(bb * L_ + l);
        const float dtv = dt[row * DI + d];
        const float xcv = xc[row * DI + d];
        const float Bv  = dbc[row * DBCP + DTR + s];
        const float Cv  = dbc[row * DBCP + DTR + DS + s];
        const float dA  = expf(dtv * Aneg);
        h = fmaf(dA, h, dtv * xcv * Bv);
        float p = h * Cv;
        p += __shfl_xor(p, 8, 32);
        p += __shfl_xor(p, 4, 32);
        p += __shfl_xor(p, 2, 32);
        p += __shfl_xor(p, 1, 32);
        if (s == 0) {
            const float zv = xz[row * (2 * DI) + DI + d];
            y[row * DI + d] = (p + xcv * Dv) * silu_f(zv);
        }
    }
}

// ------------------------------- launcher ----------------------------------
// Input leaf order = jax tree flatten (sorted dict keys) of setup_inputs():
//   params{cat, feat_enc, fuse_e, fuse_o, mixers, pos_enc}, s_coord, s_feat,
//   s_inverse, s_order, s_s_o_i.

extern "C" void kernel_launch(void* const* d_in, const int* in_sizes, int n_in,
                              void* d_out, int out_size, void* d_ws, size_t ws_size,
                              hipStream_t stream)
{
    (void)in_sizes; (void)n_in; (void)out_size; (void)ws_size;

    auto P = [&](int i) { return (const float*)d_in[i]; };
    enum {
        CAT_B = 0, CAT_W,
        FE_B1, FE_B2, FE_W1, FE_W2,                       // feat_enc
        FUE_B1, FUE_B2, FUE_LNB, FUE_LNW, FUE_W1, FUE_W2, // fuse_e
        FUO_B1, FUO_B2, FUO_LNB, FUO_LNW, FUO_W1, FUO_W2, // fuse_o
        MIX0 = 18,                                        // mixers[0] leaves
        // block: A_log, D, conv_b, conv_w, dt_b, dt_w, in_proj, ln_b, ln_w,
        //        out_proj, x_proj, then norm_f_b, norm_f_w  (13 per mixer)
        POS_B1 = 18 + 2 * 13, POS_B2, POS_W1, POS_W2,
        IN_COORD, IN_FEAT, IN_INV, IN_ORD, IN_SOI
    };
    const int* inv_idx = (const int*)d_in[IN_INV];
    const int* ord_idx = (const int*)d_in[IN_ORD];
    float* out = (float*)d_out;

    // workspace carve-up (f32)
    float* ws = (float*)d_ws;
    size_t off = 0;
    auto take = [&](size_t n) { float* p = ws + off; off += n; return p; };
    float* H1   = take((size_t)N_ * 256);           // encoder hidden
    float* CONC = take((size_t)N_ * 384);           // [feat|coord|cat]
    float* FH   = take((size_t)N_ * 256);           // fuse hidden
    float* H    = take((size_t)N_ * DM);            // running point features
    float* XSEQ = take((size_t)B_ * L_ * DM);
    float* LNX  = take((size_t)B_ * L_ * DM);
    float* XZ   = take((size_t)B_ * L_ * 2 * DI);
    float* XC   = take((size_t)B_ * L_ * DI);
    float* DBC  = take((size_t)B_ * L_ * DBCP);     // padded to 64 cols
    float* DT   = take((size_t)B_ * L_ * DI);
    float* Y    = take((size_t)B_ * L_ * DI);
    float* MOUT = take((size_t)B_ * L_ * DM);
    float* CATO = take((size_t)N_ * O_ * DM);
    float* XWP  = take((size_t)DI * DBCP);          // padded x_proj weight

    const dim3 blk256(256);
    auto grid_ew = [&](int n) { return dim3((n + 255) / 256); };
    auto grid_mm = [&](int M, int N) { return dim3(N / 64, M / 64); };
    auto grid_sm = [&](int M, int N) { return dim3((N + 15) / 16, (M + 15) / 16); };

    // ---------------- encoder ----------------
    gemm_small_kernel<1><<<grid_sm(N_, 256), blk256, 0, stream>>>(
        P(IN_FEAT), 4, P(FE_W1), 256, P(FE_B1), H1, 256, N_, 256, 4);
    gemm_wmma_kernel<0><<<grid_mm(N_, 128), blk256, 0, stream>>>(
        H1, 256, P(FE_W2), 128, P(FE_B2), CONC + 0, 384, N_, 128, 256);
    gemm_small_kernel<1><<<grid_sm(N_, 256), blk256, 0, stream>>>(
        P(IN_COORD), 3, P(POS_W1), 256, P(POS_B1), H1, 256, N_, 256, 3);
    gemm_wmma_kernel<0><<<grid_mm(N_, 128), blk256, 0, stream>>>(
        H1, 256, P(POS_W2), 128, P(POS_B2), CONC + 128, 384, N_, 128, 256);
    cat_fill_kernel<<<grid_ew(N_ * 128), blk256, 0, stream>>>(
        P(IN_SOI), P(CAT_W), P(CAT_B), CONC);
    gemm_wmma_kernel<0><<<grid_mm(N_, 128), blk256, 0, stream>>>(
        CONC, 384, P(FUE_W1), 128, P(FUE_B1), FH, 128, N_, 128, 384);
    ln_kernel<1><<<dim3(N_), blk256, 0, stream>>>(
        FH, P(FUE_LNW), P(FUE_LNB), FH, 128);
    gemm_wmma_kernel<0><<<grid_mm(N_, 256), blk256, 0, stream>>>(
        FH, 128, P(FUE_W2), 256, P(FUE_B2), H, 256, N_, 256, 128);

    // ---------------- mixers ----------------
    for (int mix = 0; mix < 2; ++mix) {
        const int base = MIX0 + 13 * mix;
        const float* A_log = P(base + 0);
        const float* Dprm  = P(base + 1);
        const float* cb    = P(base + 2);
        const float* cw    = P(base + 3);
        const float* dt_b  = P(base + 4);
        const float* dt_w  = P(base + 5);
        const float* inW   = P(base + 6);
        const float* ln_b  = P(base + 7);
        const float* ln_w  = P(base + 8);
        const float* outW  = P(base + 9);
        const float* xW    = P(base + 10);
        const float* nfb   = P(base + 11);
        const float* nfw   = P(base + 12);

        perm_fwd_kernel<<<grid_ew(O_ * N_ * DM), blk256, 0, stream>>>(
            H, ord_idx, XSEQ);
        ln_kernel<0><<<dim3(B_ * L_), blk256, 0, stream>>>(
            XSEQ, ln_w, ln_b, LNX, DM);
        // in_proj -> XZ [4096, 1024]
        gemm_wmma_kernel<0><<<grid_mm(B_ * L_, 2 * DI), blk256, 0, stream>>>(
            LNX, DM, inW, 2 * DI, nullptr, XZ, 2 * DI, B_ * L_, 2 * DI, DM);
        conv_silu_kernel<<<grid_ew(B_ * L_ * DI), blk256, 0, stream>>>(
            XZ, cw, cb, XC);
        // x_proj (weight zero-padded 48 -> 64 cols) -> DBC [4096, 64]
        pad_xproj_kernel<<<grid_ew(DI * DBCP), blk256, 0, stream>>>(xW, XWP);
        gemm_wmma_kernel<0><<<grid_mm(B_ * L_, DBCP), blk256, 0, stream>>>(
            XC, DI, XWP, DBCP, nullptr, DBC, DBCP, B_ * L_, DBCP, DI);
        // dt = softplus(DBC[:, :16] @ dt_w + dt_b)
        gemm_small_kernel<3><<<grid_sm(B_ * L_, DI), blk256, 0, stream>>>(
            DBC, DBCP, dt_w, DI, dt_b, DT, DI, B_ * L_, DI, DTR);
        scan_kernel<<<dim3(DI / 16, B_), blk256, 0, stream>>>(
            DT, XC, DBC, XZ, A_log, Dprm, Y);
        // out_proj -> MOUT
        gemm_wmma_kernel<0><<<grid_mm(B_ * L_, DM), blk256, 0, stream>>>(
            Y, DI, outW, DM, nullptr, MOUT, DM, B_ * L_, DM, DI);
        add_kernel<<<grid_ew(B_ * L_ * DM), blk256, 0, stream>>>(
            MOUT, XSEQ, MOUT, B_ * L_ * DM);
        ln_kernel<0><<<dim3(B_ * L_), blk256, 0, stream>>>(
            MOUT, nfw, nfb, MOUT, DM);
        perm_inv_kernel<<<grid_ew(O_ * N_ * DM), blk256, 0, stream>>>(
            MOUT, inv_idx, CATO);
        // fuse_o
        gemm_wmma_kernel<0><<<grid_mm(N_, DM), blk256, 0, stream>>>(
            CATO, O_ * DM, P(FUO_W1), DM, P(FUO_B1), FH, DM, N_, DM, O_ * DM);
        ln_kernel<1><<<dim3(N_), blk256, 0, stream>>>(
            FH, P(FUO_LNW), P(FUO_LNB), FH, DM);
        gemm_wmma_kernel<0><<<grid_mm(N_, DM), blk256, 0, stream>>>(
            FH, DM, P(FUO_W2), DM, P(FUO_B2), H, DM, N_, DM, DM);
        hipMemcpyAsync(out + (size_t)mix * N_ * DM, H,
                       (size_t)N_ * DM * sizeof(float),
                       hipMemcpyDeviceToDevice, stream);
    }
}
